// DeformGCN_26800595927568
// MI455X (gfx1250) — matrix operand order repaired and checked by reading, not compiled, not run on hardware
//
#include <hip/hip_runtime.h>
#include <cstdint>

// ---------------------------------------------------------------------------
// DeformGCN for MI455X (gfx1250).
// Every GCN layer is restructured as:
//   Y[:, 0:256]   = x @ Ws        (bf16 WMMA GEMM, f32 accum)
//   Y[:, 256:512] = x @ Wn        (same GEMM, weights concatenated)
//   Y[dst, 0:256] += Y[src, 256:512]   (edge scatter, atomics)
//   out = act(Y[:, 0:256] + bias)      (elementwise epilogue, bf16 out)
// using segment_sum(x[src]) @ Wn == segment_sum((x @ Wn)[src]).
// Activations live in bf16; A-tiles are staged to LDS with double-buffered
// global_load_async_to_lds_b128 (ASYNCcnt) overlapping the WMMAs.
// ---------------------------------------------------------------------------

typedef __attribute__((ext_vector_type(16))) __bf16 v16bf;
typedef __attribute__((ext_vector_type(8)))  float  v8f;

#define HID 256
#define NW  512   // Ws|Wn concatenated output width
#define NB  16    // batch
#define N0  2048
#define NE  12288
#define K0P 1504  // 1475 padded to multiple of 32

#if __has_builtin(__builtin_amdgcn_s_wait_asynccnt)
#define ASYNC_WAIT(n) __builtin_amdgcn_s_wait_asynccnt(n)
#else
#define ASYNC_WAIT(n) asm volatile("s_wait_asynccnt %0" :: "i"(n) : "memory")
#endif

__device__ __forceinline__ void async_b128_to_lds(uint32_t lds_addr, uint64_t gaddr) {
  asm volatile("global_load_async_to_lds_b128 %0, %1, off"
               :: "v"(lds_addr), "v"(gaddr) : "memory");
}

// -------------------------------- GEMM -------------------------------------
// Y[M x 512] = A[M x K] * W[K x 512], A bf16 row-major (lda = K), W bf16
// pre-swizzled into WMMA B-fragment layout:
//   off(nt,kt,n,k) = ((nt*KT + kt)*16 + n)*32 + k
// Workgroup: 256 threads = 8 waves (2 M-waves x 4 N-waves), tile 128x256.
__global__ __launch_bounds__(256)
void gemm_wmma(const __bf16* __restrict__ A, const __bf16* __restrict__ W,
               float* __restrict__ Y, int KT, int lda) {
  __shared__ __bf16 As[2][128][40];          // 32 k + pad, 16B aligned rows

  const int m0     = blockIdx.x * 128;
  const int ntbase = blockIdx.y * 16;        // 16 n-tiles per workgroup
  const int c0     = blockIdx.y * 256;

  const int lane = threadIdx.x & 31;
  const int w    = threadIdx.x >> 5;
  const int wm   = w & 1;                    // 0..1  (64 rows each)
  const int wn   = w >> 1;                   // 0..3  (64 cols each)
  const int half = lane >> 4;
  const int lm   = lane & 15;

  // per-thread staging slots: 2 x 16B chunks of the 128x32 tile
  const int q0 = threadIdx.x, q1 = threadIdx.x + 256;
  const int row0 = q0 >> 2, c80 = (q0 & 3) * 8;
  const int row1 = q1 >> 2, c81 = (q1 & 3) * 8;

  v8f acc[4][4];
  #pragma unroll
  for (int i = 0; i < 4; ++i)
    #pragma unroll
    for (int j = 0; j < 4; ++j)
      #pragma unroll
      for (int v = 0; v < 8; ++v) acc[i][j][v] = 0.0f;

  auto prefetch = [&](int kt, int b) {
    const uint64_t ga0 = (uint64_t)(A + (size_t)(m0 + row0) * lda + kt * 32 + c80);
    const uint64_t ga1 = (uint64_t)(A + (size_t)(m0 + row1) * lda + kt * 32 + c81);
    const uint32_t la0 = (uint32_t)(uintptr_t)&As[b][row0][c80];
    const uint32_t la1 = (uint32_t)(uintptr_t)&As[b][row1][c81];
    async_b128_to_lds(la0, ga0);
    async_b128_to_lds(la1, ga1);
  };

  prefetch(0, 0);

  for (int kt = 0; kt < KT; ++kt) {
    const int b = kt & 1;
    if (kt + 1 < KT) {
      prefetch(kt + 1, b ^ 1);   // overlap next tile DMA with this tile's math
      ASYNC_WAIT(2);             // in-order completion => tile kt has landed
    } else {
      ASYNC_WAIT(0);
    }
    __syncthreads();

    // ---- A fragments from LDS (two contiguous 16B chunks per lane) ----
    v16bf af[4];
    #pragma unroll
    for (int ms = 0; ms < 4; ++ms) {
      const int row = wm * 64 + ms * 16 + lm;
      union { v16bf v; uint4 u[2]; } t;
      t.u[0] = *reinterpret_cast<const uint4*>(&As[b][row][half * 16]);
      t.u[1] = *reinterpret_cast<const uint4*>(&As[b][row][half * 16 + 8]);
      af[ms] = t.v;
    }
    // ---- B fragments from pre-swizzled weights (32B contiguous/lane) ----
    v16bf bfr[4];
    #pragma unroll
    for (int ns = 0; ns < 4; ++ns) {
      const int nt = ntbase + wn * 4 + ns;
      const size_t off = (((size_t)nt * KT + kt) * 16 + lm) * 32 + half * 16;
      union { v16bf v; uint4 u[2]; } t;
      const uint4* p = reinterpret_cast<const uint4*>(W + off);
      t.u[0] = p[0];
      t.u[1] = p[1];
      bfr[ns] = t.v;
    }
    #pragma unroll
    for (int ms = 0; ms < 4; ++ms)
      #pragma unroll
      for (int ns = 0; ns < 4; ++ns)
        acc[ms][ns] = __builtin_amdgcn_wmma_f32_16x16x32_bf16(
            false, af[ms], false, bfr[ns], (short)0, acc[ms][ns], false, false);
    __syncthreads();             // all waves done reading buf b before reuse
  }

  // ---- writeback: VGPR v holds M = v + 8*half, col = lane&15 ----
  #pragma unroll
  for (int ms = 0; ms < 4; ++ms) {
    #pragma unroll
    for (int ns = 0; ns < 4; ++ns) {
      const int col   = c0 + wn * 64 + ns * 16 + lm;
      const int rbase = m0 + wm * 64 + ms * 16 + 8 * half;
      #pragma unroll
      for (int v = 0; v < 8; ++v)
        Y[(size_t)(rbase + v) * NW + col] = acc[ms][ns][v];
    }
  }
}

// ------------------------- small helper kernels ----------------------------

// Concat [verts(3) | local(960) | global(512) | zero-pad] -> bf16, K=1504
__global__ void build_x0(const float* __restrict__ verts,
                         const float* __restrict__ locf,
                         const float* __restrict__ gf,
                         __bf16* __restrict__ X0) {
  const int row = blockIdx.x;            // b*2048 + i
  const int b   = row >> 11;
  __bf16* dst = X0 + (size_t)row * K0P;
  for (int k = threadIdx.x; k < K0P; k += 256) {
    float v;
    if (k < 3)          v = verts[(size_t)row * 3 + k];
    else if (k < 963)   v = locf[(size_t)row * 960 + (k - 3)];
    else if (k < 1475)  v = gf[(size_t)b * 512 + (k - 963)];
    else                v = 0.0f;
    dst[k] = (__bf16)v;
  }
}

// Pack [Ws | Wn] (f32, K x 256 each) into bf16 WMMA B-fragment layout.
__global__ void swizzle_w(const float* __restrict__ Ws,
                          const float* __restrict__ Wn,
                          __bf16* __restrict__ dst, int Korig, int KT) {
  const int idx = blockIdx.x * 256 + threadIdx.x;  // KT*16384 total
  const int kk = idx & 31;
  const int n  = (idx >> 5) & 15;
  const int t  = idx >> 9;
  const int kt = t % KT;
  const int nt = t / KT;
  const int k  = kt * 32 + kk;
  const int c  = nt * 16 + n;
  float v = 0.0f;
  if (k < Korig)
    v = (c < 256) ? Ws[(size_t)k * 256 + c] : Wn[(size_t)k * 256 + (c - 256)];
  dst[idx] = (__bf16)v;
}

// Y[b, dst, c] += Y[b, src, 256 + c]
__global__ void aggregate(float* __restrict__ Y, const int* __restrict__ edges,
                          int n) {
  const int e = blockIdx.x, b = blockIdx.y;
  const int s = edges[e];
  const int d = edges[NE + e];
  const size_t rs = ((size_t)b * n + s) * NW + 256;
  const size_t rd = ((size_t)b * n + d) * NW;
  atomicAdd(&Y[rd + threadIdx.x], Y[rs + threadIdx.x]);
}

// mode 0: dst = Y+b ; mode 1: dst = relu(Y+b) ; mode 2: dst = 0.5*(dst + relu(Y+b))
__global__ void epilogue(const float* __restrict__ Y,
                         const float* __restrict__ bias,
                         __bf16* __restrict__ dst, int n, int dn, int mode) {
  const int row = blockIdx.x;            // b*n + i
  const int b = row / n;
  const int i = row - b * n;
  const int c = threadIdx.x;
  float v = Y[(size_t)row * NW + c] + bias[c];
  if (mode >= 1) v = fmaxf(v, 0.0f);
  const size_t di = ((size_t)b * dn + i) * HID + c;
  if (mode == 2) v = 0.5f * ((float)dst[di] + v);
  dst[di] = (__bf16)v;
}

// xnext[b, n+i, :] = H[b, i, :]
__global__ void copy_half(const __bf16* __restrict__ H,
                          __bf16* __restrict__ xnext, int n) {
  const int row = blockIdx.x;
  const int b = row / n;
  const int i = row - b * n;
  xnext[((size_t)b * 2 * n + n + i) * HID + threadIdx.x] =
      H[(size_t)row * HID + threadIdx.x];
}

// Last layer (K=256 -> 3+3), one wave per row, shuffle reduce (full f32).
__global__ void last_dot(const __bf16* __restrict__ X3,
                         const float* __restrict__ Ws,
                         const float* __restrict__ Wn,
                         float* __restrict__ Yl) {
  const int row = blockIdx.x;
  const __bf16* x = X3 + (size_t)row * HID;
  float p[6] = {0, 0, 0, 0, 0, 0};
  for (int j = 0; j < 8; ++j) {
    const int k = threadIdx.x + 32 * j;
    const float xv = (float)x[k];
    #pragma unroll
    for (int c = 0; c < 3; ++c) {
      p[c]     += xv * Ws[k * 3 + c];
      p[3 + c] += xv * Wn[k * 3 + c];
    }
  }
  #pragma unroll
  for (int o = 16; o > 0; o >>= 1)
    #pragma unroll
    for (int c = 0; c < 6; ++c) p[c] += __shfl_down(p[c], o, 32);
  if (threadIdx.x == 0) {
    #pragma unroll
    for (int c = 0; c < 6; ++c) Yl[(size_t)row * 6 + c] = p[c];
  }
}

__global__ void last_agg(float* __restrict__ Yl, const int* __restrict__ edges,
                         int N3) {
  const int idx = blockIdx.x * 256 + threadIdx.x;
  if (idx >= NE * NB * 3) return;
  const int c = idx % 3;
  const int t = idx / 3;
  const int e = t % NE;
  const int b = t / NE;
  const int s = edges[e];
  const int d = edges[NE + e];
  atomicAdd(&Yl[((size_t)b * N3 + d) * 6 + c], Yl[((size_t)b * N3 + s) * 6 + 3 + c]);
}

__global__ void final_out(const float* __restrict__ Yl,
                          const float* __restrict__ bias,
                          float* __restrict__ out, int total) {
  const int idx = blockIdx.x * 256 + threadIdx.x;
  if (idx >= total * 3) return;
  const int c = idx % 3;
  const int r = idx / 3;
  out[idx] = Yl[(size_t)r * 6 + c] + bias[c];
}

// ------------------------------- host side ---------------------------------

extern "C" void kernel_launch(void* const* d_in, const int* in_sizes, int n_in,
                              void* d_out, int out_size, void* d_ws, size_t ws_size,
                              hipStream_t stream) {
  (void)in_sizes; (void)n_in; (void)out_size; (void)ws_size;
  const float* verts = (const float*)d_in[0];
  const float* locf  = (const float*)d_in[1];
  const float* gf    = (const float*)d_in[2];
  const int*   edges = (const int*)d_in[3];
  // params flattened pytree (sorted dict keys):
  // per block (base=4+9*blk): c1_Wn, c1_Ws, c1_b, c2_Wn, c2_Ws, c2_b, rWn, rWs, rb
  // then last: Wn(31), Ws(32), b(33)

  // ---- workspace bump allocator ----
  char* ws = (char*)d_ws;
  size_t off = 0;
  auto alloc = [&](size_t bytes) -> void* {
    void* p = ws + off;
    off += (bytes + 255) & ~(size_t)255;
    return p;
  };
  __bf16* Warena = (__bf16*)alloc((size_t)(47 * 32 * 512 + 23 * 8 * 32 * 512) * 2);
  __bf16* X0     = (__bf16*)alloc((size_t)NB * N0 * K0P * 2);
  __bf16* XA     = (__bf16*)alloc((size_t)NB * 16384 * HID * 2);
  __bf16* XB     = (__bf16*)alloc((size_t)NB * 8192  * HID * 2);
  __bf16* H      = (__bf16*)alloc((size_t)NB * 8192  * HID * 2);
  __bf16* H1     = (__bf16*)alloc((size_t)NB * 8192  * HID * 2);
  float*  Y      = (float*) alloc((size_t)NB * 8192  * NW  * 4);   // also Ylast

  // ---- weight element offsets & swizzle (24 GEMM layers) ----
  size_t wel[24];
  {
    size_t acc = 0;
    for (int blk = 0; blk < 3; ++blk) {
      const int base = 4 + blk * 9;
      const float* c1Wn = (const float*)d_in[base + 0];
      const float* c1Ws = (const float*)d_in[base + 1];
      const float* c2Wn = (const float*)d_in[base + 3];
      const float* c2Ws = (const float*)d_in[base + 4];
      const float* rWn  = (const float*)d_in[base + 6];
      const float* rWs  = (const float*)d_in[base + 7];
      for (int j = 0; j < 8; ++j) {
        const int lid = blk * 8 + j;
        const int Korig = (blk == 0 && j == 0) ? 1475 : 256;
        const int KT    = (blk == 0 && j == 0) ? 47 : 8;
        const float *Wsp, *Wnp;
        if (j == 0)      { Wsp = c1Ws; Wnp = c1Wn; }
        else if (j == 7) { Wsp = c2Ws; Wnp = c2Wn; }
        else             { Wsp = rWs + (size_t)(j - 1) * 65536;
                           Wnp = rWn + (size_t)(j - 1) * 65536; }
        wel[lid] = acc;
        swizzle_w<<<KT * 64, 256, 0, stream>>>(Wsp, Wnp, Warena + acc, Korig, KT);
        acc += (size_t)KT * 32 * 512;
      }
    }
  }

  // ---- build padded bf16 input ----
  build_x0<<<NB * N0, 256, 0, stream>>>(verts, locf, gf, X0);

  // ---- three bottleneck blocks ----
  int n = N0;
  const __bf16* xin = X0;
  int lda = K0P;
  __bf16* targets[3] = {XA, XB, XA};
  for (int blk = 0; blk < 3; ++blk) {
    const int base = 4 + blk * 9;
    const float* c1b = (const float*)d_in[base + 2];
    const float* c2b = (const float*)d_in[base + 5];
    const float* rb  = (const float*)d_in[base + 8];
    const int M = NB * n;
    const dim3 gg(M / 128, 2);
    const int KT = (blk == 0) ? 47 : 8;

    // c1
    gemm_wmma<<<gg, 256, 0, stream>>>(xin, Warena + wel[blk * 8], Y, KT, lda);
    aggregate<<<dim3(NE, NB), 256, 0, stream>>>(Y, edges, n);
    epilogue<<<M, 256, 0, stream>>>(Y, c1b, H, n, n, 1);

    // residual pairs
    for (int i = 0; i < 3; ++i) {
      gemm_wmma<<<gg, 256, 0, stream>>>(H, Warena + wel[blk * 8 + 1 + i * 2], Y, 8, HID);
      aggregate<<<dim3(NE, NB), 256, 0, stream>>>(Y, edges, n);
      epilogue<<<M, 256, 0, stream>>>(Y, rb + (size_t)(i * 2 + 0) * 256, H1, n, n, 1);

      gemm_wmma<<<gg, 256, 0, stream>>>(H1, Warena + wel[blk * 8 + 2 + i * 2], Y, 8, HID);
      aggregate<<<dim3(NE, NB), 256, 0, stream>>>(Y, edges, n);
      epilogue<<<M, 256, 0, stream>>>(Y, rb + (size_t)(i * 2 + 1) * 256, H, n, n, 2);
    }

    // c2 -> first half of next x, then copy h into second half
    gemm_wmma<<<gg, 256, 0, stream>>>(H, Warena + wel[blk * 8 + 7], Y, 8, HID);
    aggregate<<<dim3(NE, NB), 256, 0, stream>>>(Y, edges, n);
    epilogue<<<M, 256, 0, stream>>>(Y, c2b, targets[blk], n, 2 * n, 0);
    copy_half<<<M, 256, 0, stream>>>(H, targets[blk], n);

    xin = targets[blk];
    lda = HID;
    n *= 2;
  }

  // ---- last GCN: (B, 16384, 256) -> (B, 16384, 3) ----
  const float* lastWn = (const float*)d_in[31];
  const float* lastWs = (const float*)d_in[32];
  const float* lastb  = (const float*)d_in[33];
  const int N3 = 16384;
  float* Yl = Y;
  last_dot<<<NB * N3, 32, 0, stream>>>(XA, lastWs, lastWn, Yl);
  {
    const int tot = NE * NB * 3;
    last_agg<<<(tot + 255) / 256, 256, 0, stream>>>(Yl, edges, N3);
  }
  {
    const int tot = NB * N3;
    final_out<<<(tot * 3 + 255) / 256, 256, 0, stream>>>(Yl, lastb, (float*)d_out, tot);
  }
}
